// Generator_33165737459979
// MI455X (gfx1250) — matrix operand order repaired and checked by reading, not compiled, not run on hardware
//
#include <hip/hip_runtime.h>
#include <hip/hip_bf16.h>

typedef __bf16 bf16_t;
typedef __attribute__((ext_vector_type(16))) __bf16 v16bf;
typedef __attribute__((ext_vector_type(8)))  __bf16 v8bf;
typedef __attribute__((ext_vector_type(8)))  float  v8f;

#define SEQL 601
#define BATCH 512
#define HID  256
#define ZD   100
#define FD   5
#define XS   520   // xh LDS row stride in bf16 elems (512 + 8 pad)
#define ZS   128   // zbuf LDS row stride in bf16 elems (padded K for fc1)
#define NT   512   // threads per workgroup (16 waves)

// ---------------- prep kernels: repack weights into WMMA B-fragment order ----------------
// B-fragment (32x16 bf16, wave32): lane L -> column n = L&15 ; element e -> k = (e&7) + (e>>3)*16 + (L<16?0:8)
// Fragment stream layout: [ntile][ktile][lane][e] contiguous bf16.

__global__ void pack_lstm_k(const float* __restrict__ w_ih, const float* __restrict__ w_hh,
                            bf16_t* __restrict__ dst) {
  int idx = blockIdx.x * 256 + threadIdx.x;          // 64 nt * 16 kt * 32 lanes * 16 e
  if (idx >= 64 * 16 * 32 * 16) return;
  int e    =  idx        & 15;
  int lane = (idx >> 4)  & 31;
  int kt   = (idx >> 9)  & 15;
  int nt   =  idx >> 13;
  int n = nt * 16 + (lane & 15);
  int k = kt * 32 + (e & 7) + ((e >> 3) << 4) + ((lane < 16) ? 0 : 8);
  float v = (k < HID) ? w_ih[n * HID + k] : w_hh[n * HID + (k - HID)];
  dst[idx] = (bf16_t)v;
}

__global__ void pack_fc1_k(const float* __restrict__ fc1_w, bf16_t* __restrict__ dst) {
  int idx = blockIdx.x * 256 + threadIdx.x;          // 16 nt * 4 kt * 32 * 16
  if (idx >= 16 * 4 * 32 * 16) return;
  int e    =  idx        & 15;
  int lane = (idx >> 4)  & 31;
  int kt   = (idx >> 9)  & 3;
  int nt   =  idx >> 11;
  int n = nt * 16 + (lane & 15);
  int k = kt * 32 + (e & 7) + ((e >> 3) << 4) + ((lane < 16) ? 0 : 8);
  float v = (k < ZD + FD) ? fc1_w[n * (ZD + FD) + k] : 0.f;
  dst[idx] = (bf16_t)v;
}

__global__ void bias_sum_k(const float* __restrict__ a, const float* __restrict__ b,
                           float* __restrict__ dst, int n) {
  int i = blockIdx.x * 256 + threadIdx.x;
  if (i < n) dst[i] = a[i] + b[i];
}

// ---------------- LSTM cell: fused gate GEMM + nonlinearity ----------------
// 16 waves: wave w owns N-tiles {w + 16p, p=0..3} == gates (i,f,g,o) of hidden tile w.

__device__ __forceinline__ void lstm_cell_step(
    const bf16_t* __restrict__ xh_in,   // [16][XS] bf16 : concat(x_or_h1, h_prev)
    const bf16_t* __restrict__ Wp,      // packed B fragments [64][16][32][16]
    const float* __restrict__ bias_r,   // 4 pre-hoisted per-lane gate biases (i,f,g,o)
    int wave, int lane, int nl, int mh,
    float cst[8],                       // per-thread cell state (C-fragment layout)
    bf16_t* __restrict__ hdst0,         // h_new dest 0 (row*XS + j)
    bf16_t* __restrict__ hdst1)         // optional h_new dest 1
{
  v8f acc[4];
#pragma unroll
  for (int p = 0; p < 4; p++) {
    v8f av;
#pragma unroll
    for (int r = 0; r < 8; r++) av[r] = bias_r[p];
    acc[p] = av;
  }

#pragma unroll 4
  for (int kt = 0; kt < 16; kt++) {
    int k0 = kt * 32 + ((lane >> 4) << 3);
    v8bf alo = *(const v8bf*)&xh_in[nl * XS + k0];
    v8bf ahi = *(const v8bf*)&xh_in[nl * XS + k0 + 16];
    v16bf afrag = __builtin_shufflevector(alo, ahi, 0,1,2,3,4,5,6,7,8,9,10,11,12,13,14,15);
#pragma unroll
    for (int p = 0; p < 4; p++) {
      int nt = wave + 16 * p;
      v16bf bfrag = *(const v16bf*)&Wp[((nt * 16 + kt) * 32 + lane) * 16];
      acc[p] = __builtin_amdgcn_wmma_f32_16x16x32_bf16(false, afrag, false, bfrag,
                                                       (short)0, acc[p], false, false);
    }
  }

  __syncthreads();  // all GEMM reads of xh_in done before overwriting h region

  const int j = wave * 16 + nl;   // hidden unit (fragment column)
#pragma unroll
  for (int r = 0; r < 8; r++) {
    float iv = acc[0][r];
    float fv = acc[1][r];
    float gv = acc[2][r];
    float ov = acc[3][r];
    iv = 1.f / (1.f + __expf(-iv));
    fv = 1.f / (1.f + __expf(-fv));
    ov = 1.f / (1.f + __expf(-ov));
    gv = tanhf(gv);
    float cn = fv * cst[r] + iv * gv;
    cst[r] = cn;
    float hn = ov * tanhf(cn);
    bf16_t hb = (bf16_t)hn;
    int row = r + mh;
    hdst0[row * XS + j] = hb;
    if (hdst1) hdst1[row * XS + j] = hb;
  }
}

// ---------------- main persistent kernel: one WG = 16 batch rows, all 601 steps ----------------

__global__ __launch_bounds__(NT)
void gen_lstm_kernel(const float* __restrict__ z,
                     const float* __restrict__ prev_gen,
                     const float* __restrict__ fc1_b,
                     const bf16_t* __restrict__ F1p,
                     const bf16_t* __restrict__ W1p,
                     const float*  __restrict__ bias1,
                     const bf16_t* __restrict__ W2p,
                     const float*  __restrict__ bias2,
                     const float* __restrict__ fc2_w,
                     const float* __restrict__ fc2_b,
                     float* __restrict__ out)
{
  __shared__ bf16_t zbuf[16 * ZS];   // [16 rows][128]: z_t (0..99), prev (100..104), zero pad
  __shared__ bf16_t xh1[16 * XS];    // [x (0..255) | h1 (256..511)]
  __shared__ bf16_t xh2[16 * XS];    // [h1 (0..255) | h2 (256..511)]

  const int tid  = threadIdx.x;
  const int wave = tid >> 5;          // 0..15
  const int lane = tid & 31;
  const int nl   = lane & 15;         // fragment column
  const int mh   = (lane >> 4) << 3;  // row offset 0 or 8
  const int b0   = blockIdx.x * 16;

  for (int i = tid; i < 16 * XS; i += NT) { xh1[i] = (bf16_t)0.f; xh2[i] = (bf16_t)0.f; }
  for (int i = tid; i < 16 * ZS; i += NT) zbuf[i] = (bf16_t)0.f;
  __syncthreads();
  if (tid < 16 * FD) {
    int r = tid / FD, j = tid % FD;
    zbuf[r * ZS + ZD + j] = (bf16_t)prev_gen[(b0 + r) * FD + j];
  }

  // ---- hoist all loop-invariant parameters into registers ----
  float b1r[4], b2r[4];
#pragma unroll
  for (int p = 0; p < 4; p++) {
    b1r[p] = bias1[(wave + 16 * p) * 16 + nl];
    b2r[p] = bias2[(wave + 16 * p) * 16 + nl];
  }
  const float fb = fc1_b[wave * 16 + nl];
  v16bf f1frag[4];
#pragma unroll
  for (int kt = 0; kt < 4; kt++)
    f1frag[kt] = *(const v16bf*)&F1p[((wave * 4 + kt) * 32 + lane) * 16];

  float c1[8], c2[8];
#pragma unroll
  for (int r = 0; r < 8; r++) { c1[r] = 0.f; c2[r] = 0.f; }

  for (int t = 0; t < SEQL; t++) {
    __syncthreads();
    // (a) stage z_t as bf16
    for (int idx = tid; idx < 16 * ZD; idx += NT) {
      int r = idx / ZD, cc = idx % ZD;
      zbuf[r * ZS + cc] = (bf16_t)z[((size_t)(b0 + r) * SEQL + t) * ZD + cc];
    }
    __syncthreads();

    // (b) fc1: x = relu([z_t, prev] @ fc1_w^T + fc1_b) -> xh1[:, 0:256]  (wave owns 1 N-tile)
    {
      v8f acc;
#pragma unroll
      for (int r = 0; r < 8; r++) acc[r] = fb;
#pragma unroll
      for (int kt = 0; kt < 4; kt++) {
        int k0 = kt * 32 + ((lane >> 4) << 3);
        v8bf alo = *(const v8bf*)&zbuf[nl * ZS + k0];
        v8bf ahi = *(const v8bf*)&zbuf[nl * ZS + k0 + 16];
        v16bf afrag = __builtin_shufflevector(alo, ahi, 0,1,2,3,4,5,6,7,8,9,10,11,12,13,14,15);
        acc = __builtin_amdgcn_wmma_f32_16x16x32_bf16(false, afrag, false, f1frag[kt],
                                                      (short)0, acc, false, false);
      }
      int col = wave * 16 + nl;
#pragma unroll
      for (int r = 0; r < 8; r++) {
        float v = acc[r];
        v = v > 0.f ? v : 0.f;
        xh1[(r + mh) * XS + col] = (bf16_t)v;
      }
    }
    __syncthreads();

    // (c/d) LSTM cell 1: gates = [x,h1] @ [w_ih1;w_hh1]^T + bias1
    lstm_cell_step(xh1, W1p, b1r, wave, lane, nl, mh, c1,
                   xh1 + HID /* h1 -> xh1[:,256:] */, xh2 /* h1 -> xh2[:,0:256] */);
    __syncthreads();

    // (e/f) LSTM cell 2
    lstm_cell_step(xh2, W2p, b2r, wave, lane, nl, mh, c2,
                   xh2 + HID /* h2 -> xh2[:,256:] */, nullptr);
    __syncthreads();

    // (g) fc2: prev = tanh(h2 @ fc2_w^T + fc2_b); emit output; feed back into zbuf
    if (tid < 16 * FD) {
      int r = tid / FD, j = tid % FD;
      const bf16_t* hrow = xh2 + r * XS + HID;
      const float*  wrow = fc2_w + j * HID;
      float acc = fc2_b[j];
#pragma unroll
      for (int kb = 0; kb < 32; kb++) {
        v8bf hv = *(const v8bf*)&hrow[kb * 8];
#pragma unroll
        for (int e = 0; e < 8; e++) acc += (float)hv[e] * wrow[kb * 8 + e];
      }
      float p = tanhf(acc);
      out[((size_t)(b0 + r) * SEQL + t) * FD + j] = p;
      zbuf[r * ZS + ZD + j] = (bf16_t)p;
    }
  }
}

// ---------------- launcher ----------------

extern "C" void kernel_launch(void* const* d_in, const int* in_sizes, int n_in,
                              void* d_out, int out_size, void* d_ws, size_t ws_size,
                              hipStream_t stream) {
  (void)in_sizes; (void)n_in; (void)out_size; (void)ws_size;
  const float* z        = (const float*)d_in[0];
  const float* prev_gen = (const float*)d_in[1];
  const float* fc1_w    = (const float*)d_in[2];
  const float* fc1_b    = (const float*)d_in[3];
  const float* w_ih1    = (const float*)d_in[4];
  const float* w_hh1    = (const float*)d_in[5];
  const float* b_ih1    = (const float*)d_in[6];
  const float* b_hh1    = (const float*)d_in[7];
  const float* w_ih2    = (const float*)d_in[8];
  const float* w_hh2    = (const float*)d_in[9];
  const float* b_ih2    = (const float*)d_in[10];
  const float* b_hh2    = (const float*)d_in[11];
  const float* fc2_w    = (const float*)d_in[12];
  const float* fc2_b    = (const float*)d_in[13];
  float* out = (float*)d_out;

  bf16_t* W1p = (bf16_t*)d_ws;          // 64*16*32*16 = 524288 bf16 (1 MB)
  bf16_t* W2p = W1p + 524288;           // 1 MB
  bf16_t* F1p = W2p + 524288;           // 16*4*32*16 = 32768 bf16 (64 KB)
  float*  bias1 = (float*)(F1p + 32768);
  float*  bias2 = bias1 + 1024;

  pack_lstm_k<<<2048, 256, 0, stream>>>(w_ih1, w_hh1, W1p);
  pack_lstm_k<<<2048, 256, 0, stream>>>(w_ih2, w_hh2, W2p);
  pack_fc1_k<<<128, 256, 0, stream>>>(fc1_w, F1p);
  bias_sum_k<<<4, 256, 0, stream>>>(b_ih1, b_hh1, bias1, 1024);
  bias_sum_k<<<4, 256, 0, stream>>>(b_ih2, b_hh2, bias2, 1024);

  gen_lstm_kernel<<<BATCH / 16, NT, 0, stream>>>(
      z, prev_gen, fc1_b, F1p, W1p, bias1, W2p, bias2, fc2_w, fc2_b, out);
}
